// GCN_layer_6_56126632624284
// MI455X (gfx1250) — compile-verified, matrix operand order checked by reading, not run on hardware
//
#include <hip/hip_runtime.h>

typedef __attribute__((ext_vector_type(16))) __bf16 v16bf;
typedef __attribute__((ext_vector_type(8)))  __bf16 v8bf;
typedef __attribute__((ext_vector_type(8)))  float  v8f;

#define N_NODES 100000
#define N_EDGES 1600000
#define K_DIM   128

// ---------------------------------------------------------------------------
// f32 -> bf16 (round-to-nearest-even), stored as raw 16-bit
// ---------------------------------------------------------------------------
__device__ __forceinline__ unsigned short f32_to_bf16_rne(float f) {
  unsigned int u = __float_as_uint(f);
  unsigned int rnd = 0x7FFFu + ((u >> 16) & 1u);
  return (unsigned short)((u + rnd) >> 16);
}

__global__ void cvt_f32_bf16_kernel(const float* __restrict__ in,
                                    unsigned short* __restrict__ out, int n) {
  int i = blockIdx.x * blockDim.x + threadIdx.x;
  if (i < n) out[i] = f32_to_bf16_rne(in[i]);
}

__global__ void relu_bf16_kernel(const float* __restrict__ agg,
                                 unsigned short* __restrict__ h, int n) {
  int i = blockIdx.x * blockDim.x + threadIdx.x;
  if (i < n) {
    float v = agg[i];
    h[i] = f32_to_bf16_rne(v > 0.0f ? v : 0.0f);
  }
}

__global__ void copy_f32_kernel(const float* __restrict__ in,
                                float* __restrict__ out, int n) {
  int i = blockIdx.x * blockDim.x + threadIdx.x;
  if (i < n) out[i] = in[i];
}

// agg[n, j] = bias[j]   (dout is a power of two)
__global__ void bias_init_kernel(const float* __restrict__ b,
                                 float* __restrict__ agg, int total, int dmask) {
  int i = blockIdx.x * blockDim.x + threadIdx.x;
  if (i < total) agg[i] = b[i & dmask];
}

// ---------------------------------------------------------------------------
// Pack W [128, dout] f32 -> per-(tile_n, kstep, lane) WMMA B fragments (bf16).
// Fragment id = (tile_n*4 + kstep)*32 + lane; each fragment = 16 bf16 = 32 B.
// Per the ISA 16-bit B layout: lane (hi,ln) holds col = 16*tile_n + ln,
// K = kstep*32 + 8*hi + {0..7, 16..23}.
// ---------------------------------------------------------------------------
__global__ void pack_w_kernel(const float* __restrict__ W,
                              unsigned short* __restrict__ Bp, int dout) {
  int t = blockIdx.x * blockDim.x + threadIdx.x;
  int n_tiles = dout >> 4;
  int total = n_tiles * 4 * 32;       // fragments * lanes
  if (t >= total) return;
  int lane   = t & 31;
  int kstep  = (t >> 5) & 3;
  int tile_n = t >> 7;                // 4*32 = 128 threads per tile_n
  int hi = lane >> 4, ln = lane & 15;
  int kb  = kstep * 32 + (hi << 3);
  int col = tile_n * 16 + ln;
  unsigned short* dst = Bp + (size_t)t * 16;
#pragma unroll
  for (int i = 0; i < 8; ++i) {
    dst[i]     = f32_to_bf16_rne(W[(size_t)(kb + i) * dout + col]);
    dst[i + 8] = f32_to_bf16_rne(W[(size_t)(kb + 16 + i) * dout + col]);
  }
}

// ---------------------------------------------------------------------------
// WMMA GEMM:  C[M, dout] = A[M,128](bf16) @ W(packed bf16), f32 accumulate.
//  - packed W (<=32KB) staged once per block into LDS; B frags via ds_load_b128
//  - one wave owns a 16-row strip: 4 A fragments preloaded, loop over n-tiles
//  - C tile bounced through per-wave LDS (stride-20 pad: conflict-free writes,
//    16B-aligned reads) then stored as 2x b128 per lane (coalesced-ish)
// ---------------------------------------------------------------------------
__global__ __launch_bounds__(256)
void gcn_wmma_gemm_kernel(const unsigned short* __restrict__ Abits,
                          const unsigned short* __restrict__ Bpackbits,
                          float* __restrict__ C, int M, int dout) {
  __shared__ alignas(32) unsigned short wlds[8 * 4 * 32 * 16]; // 32 KB max
  __shared__ float cshuf[8][320];                              // 16 rows * 20

  const int lane   = threadIdx.x & 31;
  const int wave   = threadIdx.x >> 5;
  const int n_tiles = dout >> 4;

  // ---- stage packed W into LDS (uniform across block) ----
  {
    const uint4* src = (const uint4*)Bpackbits;
    uint4* dst = (uint4*)wlds;
    int nvec = n_tiles * 4 * 32 * 2;          // 32 B per fragment = 2x uint4
    for (int i = threadIdx.x; i < nvec; i += 256) dst[i] = src[i];
  }
  __syncthreads();

  const int tile_m = blockIdx.x * 8 + wave;   // uniform across the wave
  if (tile_m * 16 >= M) return;               // whole-wave uniform exit

  const int hi = lane >> 4;                   // 0: lanes 0-15, 1: lanes 16-31
  const int ln = lane & 15;
  const int kb = hi << 3;                     // K sub-offset (ISA 16-bit layout)
  const int row = tile_m * 16 + ln;

  const __bf16* __restrict__ A = (const __bf16*)Abits;

  // Preload the 4 A fragments covering K = 0..127 for this lane's row.
  v16bf af[4];
#pragma unroll
  for (int k = 0; k < 4; ++k) {
    const __bf16* arow = A + (size_t)row * K_DIM + k * 32 + kb;
    v8bf alo = *(const v8bf*)(arow);
    v8bf ahi = *(const v8bf*)(arow + 16);
    af[k] = __builtin_shufflevector(alo, ahi,
                                    0, 1, 2, 3, 4, 5, 6, 7,
                                    8, 9, 10, 11, 12, 13, 14, 15);
  }

  const v16bf* __restrict__ Bl = (const v16bf*)wlds;
  float* __restrict__ cw = &cshuf[wave][0];

  // readback coords: lane covers 8 consecutive floats of the 16x16 tile
  const int mr  = lane >> 1;                  // row in tile
  const int mc  = (lane & 1) << 3;            // col 0 or 8

  for (int tn = 0; tn < n_tiles; ++tn) {
    v8f acc = {};
#pragma unroll
    for (int k = 0; k < 4; ++k) {
      v16bf bfr = Bl[(tn * 4 + k) * 32 + lane];   // 2x ds_load_b128
      acc = __builtin_amdgcn_wmma_f32_16x16x32_bf16(
          /*neg_a=*/false, af[k], /*neg_b=*/false, bfr,
          /*c_mod=*/(short)0, acc, /*reuse_a=*/false, /*reuse_b=*/false);
    }

    // scatter acc into per-wave LDS: element (row r+8*hi, col ln)
#pragma unroll
    for (int r = 0; r < 8; ++r) cw[(r + kb) * 20 + ln] = acc[r];
    __builtin_amdgcn_wave_barrier();            // wave-internal exchange

    // coalesced readback + b128 stores (DScnt ordering handled by compiler)
    float4 lo = *(const float4*)&cw[mr * 20 + mc];
    float4 hi4 = *(const float4*)&cw[mr * 20 + mc + 4];
    float* cptr = C + (size_t)(tile_m * 16 + mr) * dout + tn * 16 + mc;
    *(float4*)(cptr)     = lo;
    *(float4*)(cptr + 4) = hi4;
    __builtin_amdgcn_wave_barrier();            // before next tn overwrites cw
  }
}

// ---------------------------------------------------------------------------
// COO scatter-add: dout/4 lanes per edge, each lane moves a float4 (b128 load,
// 4x f32 atomics). support/agg are L2-resident (102 MB < 192 MB L2).
// ---------------------------------------------------------------------------
__global__ void edge_scatter_kernel(const int* __restrict__ src,
                                    const int* __restrict__ dst,
                                    const float* __restrict__ w,
                                    const float* __restrict__ support,
                                    float* __restrict__ agg,
                                    int lg_q) {                 // log2(dout/4)
  long t = (long)blockIdx.x * blockDim.x + threadIdx.x;
  int e = (int)(t >> lg_q);
  int q = (int)(t & ((1 << lg_q) - 1));
  if (e < N_EDGES) {
    int dout = 4 << lg_q;
    const float4 v4 = *(const float4*)(support + (size_t)src[e] * dout + q * 4);
    float wt = w[e];
    float* a = agg + (size_t)dst[e] * dout + q * 4;
    atomicAdd(a + 0, v4.x * wt);
    atomicAdd(a + 1, v4.y * wt);
    atomicAdd(a + 2, v4.z * wt);
    atomicAdd(a + 3, v4.w * wt);
  }
}

// ---------------------------------------------------------------------------
extern "C" void kernel_launch(void* const* d_in, const int* in_sizes, int n_in,
                              void* d_out, int out_size, void* d_ws, size_t ws_size,
                              hipStream_t stream) {
  const float* x    = (const float*)d_in[0];
  const int*   esrc = (const int*)d_in[1];
  const int*   edst = (const int*)d_in[2];
  const float* ew   = (const float*)d_in[3];
  const float* W[6];
  const float* bias[6];
  for (int i = 0; i < 6; ++i) {
    W[i]    = (const float*)d_in[4 + 2 * i];
    bias[i] = (const float*)d_in[5 + 2 * i];
  }

  // workspace carve-up (256B aligned regions)
  char* ws = (char*)d_ws;
  size_t off = 0;
  unsigned short* hbf = (unsigned short*)(ws + off);            // [N,128] bf16
  off += (size_t)N_NODES * K_DIM * sizeof(unsigned short);
  off = (off + 255) & ~(size_t)255;
  float* support = (float*)(ws + off);                          // [N,128] f32
  off += (size_t)N_NODES * K_DIM * sizeof(float);
  off = (off + 255) & ~(size_t)255;
  float* agg = (float*)(ws + off);                              // [N,128] f32
  off += (size_t)N_NODES * K_DIM * sizeof(float);
  off = (off + 255) & ~(size_t)255;
  unsigned short* Wpk = (unsigned short*)(ws + off);            // packed W frags

  const int thr = 256;
  // layer-0 input -> bf16
  {
    int n = N_NODES * K_DIM;
    cvt_f32_bf16_kernel<<<(n + thr - 1) / thr, thr, 0, stream>>>(x, hbf, n);
  }

  for (int layer = 0; layer < 6; ++layer) {
    const int dout = (layer == 5) ? 64 : 128;
    const int lgq  = (layer == 5) ? 4 : 5;    // log2(dout/4)

    // weights -> packed WMMA B fragments (bf16)
    {
      int n = (dout / 16) * 4 * 32;           // threads = fragments*lanes
      pack_w_kernel<<<(n + thr - 1) / thr, thr, 0, stream>>>(W[layer], Wpk, dout);
    }

    // support = H @ W  (WMMA)
    {
      int m_tiles = N_NODES / 16;             // 6250, N divisible by 16
      int blocks = (m_tiles + 7) / 8;
      gcn_wmma_gemm_kernel<<<blocks, thr, 0, stream>>>(hbf, Wpk, support,
                                                       N_NODES, dout);
    }

    // agg = broadcast(bias)
    {
      int n = N_NODES * dout;
      bias_init_kernel<<<(n + thr - 1) / thr, thr, 0, stream>>>(bias[layer], agg,
                                                                n, dout - 1);
    }

    // agg += scatter(support[src] * w -> dst)
    {
      long total = (long)N_EDGES << lgq;
      int blocks = (int)((total + thr - 1) / thr);
      edge_scatter_kernel<<<blocks, thr, 0, stream>>>(esrc, edst, ew, support,
                                                      agg, lgq);
    }

    // activation / output
    if (layer < 5) {
      int n = N_NODES * dout;
      relu_bf16_kernel<<<(n + thr - 1) / thr, thr, 0, stream>>>(agg, hbf, n);
    } else {
      int n = N_NODES * dout;
      copy_f32_kernel<<<(n + thr - 1) / thr, thr, 0, stream>>>(agg, (float*)d_out, n);
    }
  }
}
